// GraphAttentionLayer_43602507989240
// MI455X (gfx1250) — matrix operand order repaired and checked by reading, not compiled, not run on hardware
//
#include <hip/hip_runtime.h>
#include <stdint.h>

// ---------------- CDNA5 WMMA plumbing (gfx1250, wave32) ----------------
typedef __attribute__((ext_vector_type(16))) _Float16 v16h;
typedef __attribute__((ext_vector_type(8)))  _Float16 v8h;
typedef __attribute__((ext_vector_type(8)))  float    v8f;

#define NROWS 8192
#define FDIM  256
#define ALPHA 0.2f
#define KC    64      // K-chunk staged in LDS per iteration
#define PADH  80      // LDS row stride in halves (160B, 16B aligned)

// A fragment: 16x32 f16 (MxK). lane m = L&15, hi = L>>4.
__device__ __forceinline__ v16h ld_a_frag(const _Float16* rowp, int hi) {
    v8h lo = *(const v8h*)(rowp + hi * 8);
    v8h hh = *(const v8h*)(rowp + 16 + hi * 8);
    return __builtin_shufflevector(lo, hh, 0,1,2,3,4,5,6,7,8,9,10,11,12,13,14,15);
}
// B fragment: 32x16 f16 (KxN), staged K-contiguous per output column n.
__device__ __forceinline__ v16h ld_b_frag(const _Float16* rowp, int hi) {
    v8h lo = *(const v8h*)(rowp + hi * 8);
    v8h hh = *(const v8h*)(rowp + 16 + hi * 8);
    return __builtin_shufflevector(lo, hh, 0,1,2,3,4,5,6,7,8,9,10,11,12,13,14,15);
}
__device__ __forceinline__ v8f wmma_f16(v16h a, v16h b, v8f c) {
    return __builtin_amdgcn_wmma_f32_16x16x32_f16(false, a, false, b, (short)0, c, false, false);
}
union PK2 { _Float16 h[2]; uint32_t u; };
union PK4 { _Float16 h[4]; uint2 u2; };

__device__ __forceinline__ float lrelu(float s) { return (s > 0.f) ? s : ALPHA * s; }

// ---------------- Kernel 1: seq = inp@Wt + bt ; f1,f2 ; seqT(f16) ----------------
__global__ __launch_bounds__(256)
void k_transform(const float* __restrict__ inp, const float* __restrict__ Wt,
                 const float* __restrict__ bt, const float* __restrict__ w1,
                 const float* __restrict__ b1, const float* __restrict__ w2,
                 const float* __restrict__ b2,
                 _Float16* __restrict__ seqT,
                 float* __restrict__ f1g, float* __restrict__ f2g)
{
    __shared__ __align__(16) _Float16 at[16][PADH];
    __shared__ __align__(16) _Float16 btile[FDIM][PADH];   // Wt chunk transposed [n][k]
    __shared__ float f1s[16], f2s[16];

    const int tid  = threadIdx.x;
    const int lane = tid & 31;
    const int wv   = tid >> 5;
    const int row0 = blockIdx.x * 16;
    const int hi   = lane >> 4;
    const int n0   = wv * 32 + (lane & 15);
    const int n1   = n0 + 16;

    if (tid < 16) { f1s[tid] = b1[0]; f2s[tid] = b2[0]; }

    v8f acc0 = {}; v8f acc1 = {};

    for (int kc = 0; kc < FDIM; kc += KC) {
        __syncthreads();
        { // stage A (inp rows, f32->f16), vectorized: float4 in, uint2 out
            int i = tid >> 4, c4 = (tid & 15) * 4;
            float4 v4 = *(const float4*)(inp + (size_t)(row0 + i) * FDIM + kc + c4);
            PK4 pk; pk.h[0] = (_Float16)v4.x; pk.h[1] = (_Float16)v4.y;
                    pk.h[2] = (_Float16)v4.z; pk.h[3] = (_Float16)v4.w;
            *(uint2*)&at[i][c4] = pk.u2;
        }
        // stage B: Wt[k][n] -> btile[n][k], packed pairs of k
        for (int kk = 0; kk < KC; kk += 2) {
            float x = Wt[(size_t)(kc + kk)     * FDIM + tid];
            float y = Wt[(size_t)(kc + kk + 1) * FDIM + tid];
            PK2 pk; pk.h[0] = (_Float16)x; pk.h[1] = (_Float16)y;
            *(uint32_t*)&btile[tid][kk] = pk.u;
        }
        __syncthreads();
        #pragma unroll
        for (int ks = 0; ks < KC; ks += 32) {
            v16h a  = ld_a_frag(&at[lane & 15][ks], hi);
            v16h b0 = ld_b_frag(&btile[n0][ks], hi);
            v16h b1f= ld_b_frag(&btile[n1][ks], hi);
            acc0 = wmma_f16(a, b0, acc0);
            acc1 = wmma_f16(a, b1f, acc1);
        }
    }

    const float bt0 = bt[n0], bt1 = bt[n1];
    const float w10 = w1[n0], w11 = w1[n1];
    const float w20 = w2[n0], w21 = w2[n1];
    __syncthreads();
    #pragma unroll
    for (int r = 0; r < 8; ++r) {
        int m   = hi * 8 + r;
        int row = row0 + m;
        float v0 = acc0[r] + bt0;
        float v1 = acc1[r] + bt1;
        seqT[(size_t)n0 * NROWS + row] = (_Float16)v0;
        seqT[(size_t)n1 * NROWS + row] = (_Float16)v1;
        atomicAdd(&f1s[m], v0 * w10 + v1 * w11);   // ds_add_f32
        atomicAdd(&f2s[m], v0 * w20 + v1 * w21);
    }
    __syncthreads();
    if (tid < 16) { f1g[row0 + tid] = f1s[tid]; f2g[row0 + tid] = f2s[tid]; }
}

// ---------------- Kernel 2a: per-row softmax max & sum (1 adj pass) ----------------
__global__ __launch_bounds__(256)
void k_softmax_stats(const float* __restrict__ adj, const float* __restrict__ f1g,
                     const float* __restrict__ f2g,
                     float* __restrict__ mrow, float* __restrict__ invl)
{
    __shared__ float sm[256], sl[256];
    const int i = blockIdx.x, tid = threadIdx.x;
    const float f1i = f1g[i];
    const float* arow = adj + (size_t)i * NROWS;
    float m = -3.0e38f, l = 0.f;
    for (int j4 = tid * 4; j4 < NROWS; j4 += 1024) {
        __builtin_prefetch(arow + j4 + 1024, 0, 0);   // global_prefetch_b8
        float4 a4 = *(const float4*)(arow + j4);
        float4 f4 = *(const float4*)(f2g + j4);
        float s0 = lrelu(f1i + f4.x) + a4.x;
        float s1 = lrelu(f1i + f4.y) + a4.y;
        float s2 = lrelu(f1i + f4.z) + a4.z;
        float s3 = lrelu(f1i + f4.w) + a4.w;
        float m4 = fmaxf(fmaxf(s0, s1), fmaxf(s2, s3));
        float nm = fmaxf(m, m4);
        l = l * __expf(m - nm)
          + __expf(s0 - nm) + __expf(s1 - nm) + __expf(s2 - nm) + __expf(s3 - nm);
        m = nm;
    }
    sm[tid] = m; sl[tid] = l;
    __syncthreads();
    for (int off = 128; off > 0; off >>= 1) {
        if (tid < off) {
            float m2 = sm[tid + off], l2 = sl[tid + off];
            float ma = sm[tid],      la = sl[tid];
            float nm = fmaxf(ma, m2);
            sm[tid] = nm;
            sl[tid] = la * __expf(ma - nm) + l2 * __expf(m2 - nm);
        }
        __syncthreads();
    }
    if (tid == 0) { mrow[i] = sm[0]; invl[i] = 1.0f / sl[0]; }
}

// ---------------- Kernel 2b: ret = softmax(scores) @ seq (flash + WMMA) ----------------
__global__ __launch_bounds__(256)
void k_aggregate(const float* __restrict__ adj, const float* __restrict__ f1g,
                 const float* __restrict__ f2g, const float* __restrict__ mrow,
                 const float* __restrict__ invl, const _Float16* __restrict__ seqT,
                 float* __restrict__ ret)
{
    __shared__ __align__(16) _Float16 pt[16][PADH];     // probability tile (A)
    __shared__ __align__(16) _Float16 st[FDIM][PADH];   // seq tile, [n][k]  (B)

    const int tid  = threadIdx.x;
    const int lane = tid & 31;
    const int wv   = tid >> 5;
    const int row0 = blockIdx.x * 16;
    const int hi   = lane >> 4;
    const int n0   = wv * 32 + (lane & 15);
    const int n1   = n0 + 16;

    const int pi  = tid >> 4;
    const int pj4 = (tid & 15) * 4;
    const float f1i = f1g[row0 + pi];
    const float mi  = mrow[row0 + pi];
    const float li  = invl[row0 + pi];
    const float* arow = adj + (size_t)(row0 + pi) * NROWS;

    // async-copy addressing for the seq tile: 8 x b128 per thread
    const uint32_t stbase = (uint32_t)(size_t)&st[0][0];       // LDS byte offset (low 32b of generic)
    const uint32_t kb     = (uint32_t)(tid & 7) * 16;          // byte offset within a 128B row
    const int      nbase  = tid >> 3;                          // n row, step 32 per q
    const uint64_t gbase  = (uint64_t)(size_t)seqT;

    v8f acc0 = {}; v8f acc1 = {};

    for (int jc = 0; jc < NROWS; jc += KC) {
        __syncthreads();
        // 1) kick off async global->LDS copy of the seq tile (ASYNCcnt-tracked)
        #pragma unroll
        for (int q = 0; q < 8; ++q) {
            int n = nbase + 32 * q;
            uint32_t laddr = stbase + (uint32_t)n * (PADH * 2) + kb;
            uint64_t gaddr = gbase + (((uint64_t)n * NROWS + (uint64_t)jc) * 2) + kb;
            asm volatile("global_load_async_to_lds_b128 %0, %1, off"
                         :: "v"(laddr), "v"(gaddr) : "memory");
        }
        // 2) overlap: compute probability tile p = exp(score - m) / l   (f16, A-matrix)
        {
            __builtin_prefetch(arow + jc + KC + pj4, 0, 0);
            float4 a4 = *(const float4*)(arow + jc + pj4);
            float4 f4 = *(const float4*)(f2g + jc + pj4);
            float s0 = lrelu(f1i + f4.x) + a4.x;
            float s1 = lrelu(f1i + f4.y) + a4.y;
            float s2 = lrelu(f1i + f4.z) + a4.z;
            float s3 = lrelu(f1i + f4.w) + a4.w;
            PK4 pk;
            pk.h[0] = (_Float16)(__expf(s0 - mi) * li);
            pk.h[1] = (_Float16)(__expf(s1 - mi) * li);
            pk.h[2] = (_Float16)(__expf(s2 - mi) * li);
            pk.h[3] = (_Float16)(__expf(s3 - mi) * li);
            *(uint2*)&pt[pi][pj4] = pk.u2;
        }
        // 3) drain async copies, then make tile visible to all waves
        asm volatile("s_wait_asynccnt 0" ::: "memory");
        __syncthreads();
        // 4) WMMA accumulate: 2 K-steps x 2 column tiles per wave
        #pragma unroll
        for (int ks = 0; ks < KC; ks += 32) {
            v16h a  = ld_a_frag(&pt[lane & 15][ks], hi);
            v16h b0 = ld_b_frag(&st[n0][ks], hi);
            v16h b1f= ld_b_frag(&st[n1][ks], hi);
            acc0 = wmma_f16(a, b0, acc0);
            acc1 = wmma_f16(a, b1f, acc1);
        }
    }
    #pragma unroll
    for (int r = 0; r < 8; ++r) {
        int row = row0 + hi * 8 + r;
        ret[(size_t)row * FDIM + n0] = acc0[r];
        ret[(size_t)row * FDIM + n1] = acc1[r];
    }
}

// ---------------- Kernel 3: gated update (WMMA over K=512) ----------------
__global__ __launch_bounds__(256)
void k_gate(const float* __restrict__ ret, const float* __restrict__ inp,
            const float* __restrict__ Wg, const float* __restrict__ bg,
            float* __restrict__ out)
{
    __shared__ __align__(16) _Float16 at[16][PADH];
    __shared__ __align__(16) _Float16 btile[FDIM][PADH];

    const int tid  = threadIdx.x;
    const int lane = tid & 31;
    const int wv   = tid >> 5;
    const int row0 = blockIdx.x * 16;
    const int hi   = lane >> 4;
    const int n0   = wv * 32 + (lane & 15);
    const int n1   = n0 + 16;

    v8f acc0 = {}; v8f acc1 = {};

    for (int kc = 0; kc < 2 * FDIM; kc += KC) {
        __syncthreads();
        { // stage A = concat(ret, inp) rows (chunk is uniformly ret or inp)
            int i = tid >> 4, c4 = (tid & 15) * 4;
            const float* srcb = (kc < FDIM)
                ? ret + (size_t)(row0 + i) * FDIM + kc
                : inp + (size_t)(row0 + i) * FDIM + (kc - FDIM);
            float4 v4 = *(const float4*)(srcb + c4);
            PK4 pk; pk.h[0] = (_Float16)v4.x; pk.h[1] = (_Float16)v4.y;
                    pk.h[2] = (_Float16)v4.z; pk.h[3] = (_Float16)v4.w;
            *(uint2*)&at[i][c4] = pk.u2;
        }
        for (int kk = 0; kk < KC; kk += 2) {
            float x = Wg[(size_t)(kc + kk)     * FDIM + tid];
            float y = Wg[(size_t)(kc + kk + 1) * FDIM + tid];
            PK2 pk; pk.h[0] = (_Float16)x; pk.h[1] = (_Float16)y;
            *(uint32_t*)&btile[tid][kk] = pk.u;
        }
        __syncthreads();
        #pragma unroll
        for (int ks = 0; ks < KC; ks += 32) {
            v16h a  = ld_a_frag(&at[lane & 15][ks], hi);
            v16h b0 = ld_b_frag(&btile[n0][ks], hi);
            v16h b1f= ld_b_frag(&btile[n1][ks], hi);
            acc0 = wmma_f16(a, b0, acc0);
            acc1 = wmma_f16(a, b1f, acc1);
        }
    }
    const float bg0 = bg[n0], bg1 = bg[n1];
    #pragma unroll
    for (int r = 0; r < 8; ++r) {
        int row = row0 + hi * 8 + r;
        float u0 = 1.f / (1.f + __expf(-(acc0[r] + bg0)));
        float u1 = 1.f / (1.f + __expf(-(acc1[r] + bg1)));
        float r0 = ret[(size_t)row * FDIM + n0], x0 = inp[(size_t)row * FDIM + n0];
        float r1 = ret[(size_t)row * FDIM + n1], x1 = inp[(size_t)row * FDIM + n1];
        out[(size_t)row * FDIM + n0] = u0 * r0 + (1.f - u0) * x0;
        out[(size_t)row * FDIM + n1] = u1 * r1 + (1.f - u1) * x1;
    }
}

// ---------------- Launch ----------------
extern "C" void kernel_launch(void* const* d_in, const int* in_sizes, int n_in,
                              void* d_out, int out_size, void* d_ws, size_t ws_size,
                              hipStream_t stream) {
    (void)in_sizes; (void)n_in; (void)out_size; (void)ws_size;
    const float* inp = (const float*)d_in[0];
    const float* adj = (const float*)d_in[1];
    const float* Wt  = (const float*)d_in[2];
    const float* bt  = (const float*)d_in[3];
    const float* w1  = (const float*)d_in[4];
    const float* b1  = (const float*)d_in[5];
    const float* w2  = (const float*)d_in[6];
    const float* b2  = (const float*)d_in[7];
    const float* Wg  = (const float*)d_in[8];
    const float* bg  = (const float*)d_in[9];
    float* out = (float*)d_out;

    // workspace layout (needs ~12.2 MB)
    char* ws = (char*)d_ws;
    float*     ret  = (float*)ws;                        // 8192*256*4 = 8 MB
    _Float16*  seqT = (_Float16*)(ws + (8u << 20));      // 256*8192*2 = 4 MB
    float*     f1g  = (float*)(ws + (12u << 20));        // 32 KB each
    float*     f2g  = f1g + NROWS;
    float*     mrow = f2g + NROWS;
    float*     invl = mrow + NROWS;

    k_transform    <<<NROWS / 16, 256, 0, stream>>>(inp, Wt, bt, w1, b1, w2, b2, seqT, f1g, f2g);
    k_softmax_stats<<<NROWS,      256, 0, stream>>>(adj, f1g, f2g, mrow, invl);
    k_aggregate    <<<NROWS / 16, 256, 0, stream>>>(adj, f1g, f2g, mrow, invl, seqT, ret);
    k_gate         <<<NROWS / 16, 256, 0, stream>>>(ret, inp, Wg, bg, out);
}
